// MPSPeriodic_59880434041478
// MI455X (gfx1250) — matrix operand order repaired and checked by reading, not compiled, not run on hardware
//
#include <hip/hip_runtime.h>

typedef __attribute__((ext_vector_type(16))) _Float16 v16h;
typedef __attribute__((ext_vector_type(8)))  float    v8f;

#define NSITES 64
#define DLOC   2
#define BD     64
#define LSTR   72   // LDS row stride in halves (144 B: 16B-aligned, bank-friendly)

#if defined(__AMDGCN__) && __has_builtin(__builtin_amdgcn_global_load_async_to_lds_b128)
#define HAS_ASYNC_LDS 1
#else
#define HAS_ASYNC_LDS 0
#if defined(__AMDGCN__)
#warning "gfx1250 __builtin_amdgcn_global_load_async_to_lds_b128 not available; using VGPR-staged fallback"
#endif
#endif

// exact pointer types the async builtin expects:
//   param0: vector_size(16) int, address_space(1)  (prints as "__device__")
//   param1: vector_size(16) int, address_space(3)  (prints as "__shared__")
typedef int v4i_gcc __attribute__((vector_size(16)));
typedef __attribute__((address_space(1))) v4i_gcc* g_v4i_ptr;
typedef __attribute__((address_space(3))) v4i_gcc* l_v4i_ptr;

// tensors = param + I, cast to f16
__global__ void mps_prep_kernel(const float* __restrict__ p,
                                _Float16* __restrict__ o, int n) {
  int i = blockIdx.x * blockDim.x + threadIdx.x;
  if (i >= n) return;
  int col = i & (BD - 1);
  int row = (i >> 6) & (BD - 1);
  float v = p[i] + ((row == col) ? 1.0f : 0.0f);
  o[i] = (_Float16)v;
}

// async DMA of one 64x64 f16 site matrix (8 KB) into LDS; 256 threads x 2 x 16B
__device__ __forceinline__ void load_site_async(const _Float16* __restrict__ Tq,
                                                _Float16* dst, int tid) {
#if HAS_ASYNC_LDS
  const uint4* g = (const uint4*)Tq;
  #pragma unroll
  for (int c = tid; c < BD * 8; c += 256) {
    int r = c >> 3, cc = c & 7;
    __builtin_amdgcn_global_load_async_to_lds_b128(
        (g_v4i_ptr)(g + c), (l_v4i_ptr)&dst[r * LSTR + cc * 8], 0, 0);
  }
#else
  (void)Tq; (void)dst; (void)tid;
#endif
}

__device__ __forceinline__ void wait_async0() {
#if defined(__AMDGCN__) && __has_builtin(__builtin_amdgcn_s_wait_asynccnt)
  __builtin_amdgcn_s_wait_asynccnt(0);
#elif HAS_ASYNC_LDS
  asm volatile("s_wait_asynccnt 0" ::: "memory");
#endif
}

__launch_bounds__(256)
__global__ void mps_chain_kernel(const int* __restrict__ x,
                                 const _Float16* __restrict__ T,
                                 float* __restrict__ out, int batch) {
  __shared__ __align__(16) _Float16 eb0[BD * LSTR];  // edge ping
  __shared__ __align__(16) _Float16 eb1[BD * LSTR];  // edge pong
  __shared__ __align__(16) _Float16 lA0[BD * LSTR];  // site matrix buffer 0
  __shared__ __align__(16) _Float16 lA1[BD * LSTR];  // site matrix buffer 1
  __shared__ int   xs[NSITES];
  __shared__ float red[BD];

  const int b = blockIdx.x;
  if (b >= batch) return;
  const int tid  = threadIdx.x;
  const int lane = tid & 31;
  const int wave = tid >> 5;
  const int grp  = lane >> 4;   // half-wave select (0/1)
  const int ln16 = lane & 15;

  if (tid < NSITES) xs[tid] = x[b * NSITES + tid];
  __syncthreads();

#if HAS_ASYNC_LDS
  // kick off DMA for site 0 before doing anything else
  load_site_async(T + (size_t)(0 * DLOC + xs[0]) * BD * BD, lA0, tid);
#endif

  // edge = identity (f16) — overlaps with the site-0 DMA
  for (int i = tid; i < BD * BD; i += 256) {
    int r = i >> 6, c = i & (BD - 1);
    eb0[r * LSTR + c] = (_Float16)((r == c) ? 1.0f : 0.0f);
  }

  // 16 output tiles (4x4 grid of 16x16); each wave owns 2 tiles sharing a row
  const int ti  = wave >> 1;        // row tile 0..3
  const int tj0 = (wave & 1) * 2;   // col tiles {0,1} or {2,3}
  const int tj1 = tj0 + 1;

  for (int s = 0; s < NSITES; ++s) {
    const _Float16* __restrict__ ein  = (s & 1) ? eb1 : eb0;
    _Float16*       __restrict__ eout = (s & 1) ? eb0 : eb1;
    _Float16* bufc = (s & 1) ? lA1 : lA0;
    _Float16* bufn = (s & 1) ? lA0 : lA1;

#if HAS_ASYNC_LDS
    wait_async0();        // this wave's DMA into bufc complete
    __syncthreads();      // all waves' DMA published; bufn reads (site s-1) done
    if (s + 1 < NSITES) { // prefetch next site matrix into the other buffer
      load_site_async(T + (size_t)((s + 1) * DLOC + xs[s + 1]) * BD * BD,
                      bufn, tid);
    }
#else
    // synchronous cooperative load of selected 64x64 f16 matrix into LDS
    {
      const uint4* __restrict__ Ag =
          (const uint4*)(T + (size_t)(s * DLOC + xs[s]) * BD * BD);
      #pragma unroll
      for (int c = tid; c < BD * 8; c += 256) {
        int r = c >> 3, cc = c & 7;
        *(uint4*)&bufc[r * LSTR + cc * 8] = Ag[c];
      }
      if (s + 1 < NSITES) {
        const _Float16* nA = T + (size_t)((s + 1) * DLOC + xs[s + 1]) * BD * BD;
        __builtin_prefetch(nA + tid * 16, 0, 1);
      }
    }
    __syncthreads();
#endif

    v8f acc0 = {}, acc1 = {};
    #pragma unroll
    for (int kk = 0; kk < BD; kk += 32) {
      union { v16h v; uint4 q2[2]; } a, b0, b1;
      // A fragment (edge tile, 16x32 f16): lane = M row; halves 0..7 -> K=kk+8*grp..,
      // halves 8..15 -> K=kk+16+8*grp.. : two contiguous 16B runs
      const int M = ti * 16 + ln16;
      a.q2[0] = *(const uint4*)&ein[M * LSTR + kk + 8 * grp];
      a.q2[1] = *(const uint4*)&ein[M * LSTR + kk + 16 + 8 * grp];
      // B fragment (site matrix tile, 32x16 f16): lane = N col, halves = K in same order as A
      #pragma unroll
      for (int e = 0; e < 16; ++e) {
        const int K = kk + ((e < 8) ? (8 * grp + e) : (16 + 8 * grp + (e - 8)));
        b0.v[e] = bufc[K * LSTR + tj0 * 16 + ln16];
        b1.v[e] = bufc[K * LSTR + tj1 * 16 + ln16];
      }

      acc0 = __builtin_amdgcn_wmma_f32_16x16x32_f16(
          false, a.v, false, b0.v, (short)0, acc0, false, false);
      acc1 = __builtin_amdgcn_wmma_f32_16x16x32_f16(
          false, a.v, false, b1.v, (short)0, acc1, false, false);
    }

    // C/D layout: VGPR v -> M = ti*16 + v + 8*grp, N = tj*16 + (lane&15)
    #pragma unroll
    for (int v = 0; v < 8; ++v) {
      const int Mo = ti * 16 + v + 8 * grp;
      eout[Mo * LSTR + tj0 * 16 + ln16] = (_Float16)acc0[v];
      eout[Mo * LSTR + tj1 * 16 + ln16] = (_Float16)acc1[v];
    }
    __syncthreads();
  }

  // after 64 sites result sits in eb0; psi = trace(edge)
  if (tid < BD) red[tid] = (float)eb0[tid * LSTR + tid];
  __syncthreads();
  if (tid == 0) {
    float psi = 0.f;
    #pragma unroll
    for (int i = 0; i < BD; ++i) psi += red[i];
    out[2 * b]     = logf(fabsf(psi));                       // Re log(psi)
    out[2 * b + 1] = (psi < 0.f) ? 3.14159265358979f : 0.f;  // Im log(psi)
  }
}

extern "C" void kernel_launch(void* const* d_in, const int* in_sizes, int n_in,
                              void* d_out, int out_size, void* d_ws, size_t ws_size,
                              hipStream_t stream) {
  const int*   x  = (const int*)d_in[0];       // [BATCH, 64] int32
  const float* tp = (const float*)d_in[1];     // [64, 2, 64, 64] float32
  const int nT    = in_sizes[1];               // 524288 elements (1 MB as f16)
  const int batch = in_sizes[0] / NSITES;      // 4096

  _Float16* Tf16 = (_Float16*)d_ws;            // f16 tensor table in workspace

  mps_prep_kernel<<<(nT + 255) / 256, 256, 0, stream>>>(tp, Tf16, nT);
  mps_chain_kernel<<<batch, 256, 0, stream>>>(x, Tf16, (float*)d_out, batch);
}
